// PolicyHead_63101659513134
// MI455X (gfx1250) — matrix operand order repaired and checked by reading, not compiled, not run on hardware
//
#include <hip/hip_runtime.h>
#include <hip/hip_bf16.h>
#include <math.h>

#define BATCH   1024
#define SEQ     256
#define DMODEL  256
#define NMOVES  480
#define KDIM    512   // 2*DMODEL
#define HDIM    256   // d_hidden

#define MBLK     64          // rows per workgroup
#define NWAVES   16          // one 16-col N-tile per wave -> 16*16 = 256 = HDIM
#define TPB      (NWAVES*32) // 512 threads
#define RSTRIDE  520         // LDS row stride in bf16 elems (512 + 8 pad -> bank-conflict-free)

typedef __attribute__((ext_vector_type(16))) __bf16 v16bf;
typedef __attribute__((ext_vector_type(8)))  float  v8f;

union Frag {
    uint4  u4[2];
    v16bf  v;
};

__device__ __forceinline__ unsigned short f32_to_bf16_rne(float x) {
    unsigned u = __float_as_uint(x);
    unsigned r = u + 0x7FFFu + ((u >> 16) & 1u);
    return (unsigned short)(r >> 16);
}
__device__ __forceinline__ float bf16_bits_to_f32(unsigned short h) {
    return __uint_as_float(((unsigned)h) << 16);
}
// split two floats into packed-bf16 hi words and lo (residual) words
__device__ __forceinline__ void split2(float a, float b, unsigned &hi, unsigned &lo) {
    unsigned short ha = f32_to_bf16_rne(a);
    unsigned short hb = f32_to_bf16_rne(b);
    float ra = a - bf16_bits_to_f32(ha);
    float rb = b - bf16_bits_to_f32(hb);
    unsigned short la = f32_to_bf16_rne(ra);
    unsigned short lb = f32_to_bf16_rne(rb);
    hi = (unsigned)ha | ((unsigned)hb << 16);
    lo = (unsigned)la | ((unsigned)lb << 16);
}

// W1 (k,n) f32  ->  W1T hi/lo (n,k) bf16 in workspace (tiny; once per launch)
__global__ void prep_w1_kernel(const float* __restrict__ W1,
                               unsigned short* __restrict__ Whi,
                               unsigned short* __restrict__ Wlo) {
    int idx = blockIdx.x * blockDim.x + threadIdx.x;
    if (idx >= KDIM * HDIM) return;
    int k = idx / HDIM;
    int n = idx - k * HDIM;
    float w = W1[idx];
    unsigned short h = f32_to_bf16_rne(w);
    float r = w - bf16_bits_to_f32(h);
    unsigned short l = f32_to_bf16_rne(r);
    Whi[(size_t)n * KDIM + k] = h;
    Wlo[(size_t)n * KDIM + k] = l;
}

__global__ __launch_bounds__(TPB)
void policy_head_kernel(const float* __restrict__ emb,
                        const int*   __restrict__ sq1,
                        const int*   __restrict__ sq2,
                        const unsigned short* __restrict__ W1Thi,
                        const unsigned short* __restrict__ W1Tlo,
                        const float* __restrict__ b1,
                        const float* __restrict__ W2,
                        const float* __restrict__ b2,
                        float* __restrict__ out) {
    extern __shared__ uint4 smem4[];
    unsigned short* Xhi = (unsigned short*)smem4;                      // MBLK*RSTRIDE bf16
    unsigned short* Xlo = Xhi + MBLK * RSTRIDE;                        // MBLK*RSTRIDE bf16
    float* partial = (float*)(Xlo + MBLK * RSTRIDE);                   // MBLK floats

    const int tid  = threadIdx.x;
    const int row0 = blockIdx.x * MBLK;

    // ---- Stage 1: gather X block (concat sq1/sq2 rows), split to bf16 hi/lo in LDS
    // MBLK*KDIM/4 = 8192 float4 chunks; 16 per thread; float4 never straddles the concat seam
#pragma unroll
    for (int it = 0; it < (MBLK * KDIM / 4) / TPB; ++it) {
        int idx4 = it * TPB + tid;
        int r    = idx4 >> 7;            // 128 float4 per row
        int kq   = (idx4 & 127) << 2;    // element offset within [0,512)
        int row  = row0 + r;
        int b    = row / NMOVES;
        int mv   = row - b * NMOVES;
        const float* src;
        if (kq < DMODEL) {
            src = emb + ((size_t)(b * SEQ + sq1[mv]) * DMODEL + kq);
        } else {
            src = emb + ((size_t)(b * SEQ + sq2[mv]) * DMODEL + (kq - DMODEL));
        }
        float4 v = *(const float4*)src;
        uint2 hp, lp;
        split2(v.x, v.y, hp.x, lp.x);
        split2(v.z, v.w, hp.y, lp.y);
        *(uint2*)(Xhi + r * RSTRIDE + kq) = hp;
        *(uint2*)(Xlo + r * RSTRIDE + kq) = lp;
    }
    __syncthreads();
    if (tid < MBLK) partial[tid] = 0.0f;
    __syncthreads();

    // ---- Stage 2: bf16x3 compensated WMMA GEMM (K = 512)
    const int wave  = tid >> 5;
    const int lane  = tid & 31;
    const int lane15 = lane & 15;
    const int khalf = lane >> 4;               // 0: lanes 0-15, 1: lanes 16-31
    const int col   = (wave << 4) + lane15;    // hidden column owned by this lane

    v8f acc[4];
#pragma unroll
    for (int m = 0; m < 4; ++m) acc[m] = (v8f)0.0f;

    const unsigned short* bhiP = W1Thi + (size_t)col * KDIM;
    const unsigned short* bloP = W1Tlo + (size_t)col * KDIM;

    for (int k0 = 0; k0 < KDIM; k0 += 32) {
        // B fragment: lane<16 holds K=k0..k0+15 of its column; lane>=16 holds K=k0+16..k0+31
        int kb_b = k0 + (khalf << 4);
        Frag bhi, blo;
        const uint4* pbh = (const uint4*)(bhiP + kb_b);
        bhi.u4[0] = pbh[0]; bhi.u4[1] = pbh[1];
        const uint4* pbl = (const uint4*)(bloP + kb_b);
        blo.u4[0] = pbl[0]; blo.u4[1] = pbl[1];

        // A fragment: lane<16 holds K=k0..k0+15; lane>=16 holds K=k0+16..k0+31
        int kb_a = k0 + (khalf << 4);
#pragma unroll
        for (int m = 0; m < 4; ++m) {
            int arow = (m << 4) + lane15;
            const unsigned short* aph = Xhi + arow * RSTRIDE + kb_a;
            const unsigned short* apl = Xlo + arow * RSTRIDE + kb_a;
            Frag ahi, alo;
            ahi.u4[0] = *(const uint4*)aph;
            ahi.u4[1] = *(const uint4*)(aph + 8);
            alo.u4[0] = *(const uint4*)apl;
            alo.u4[1] = *(const uint4*)(apl + 8);

            acc[m] = __builtin_amdgcn_wmma_f32_16x16x32_bf16(
                false, ahi.v, false, bhi.v, (short)0, acc[m], false, false);
            acc[m] = __builtin_amdgcn_wmma_f32_16x16x32_bf16(
                false, alo.v, false, bhi.v, (short)0, acc[m], false, false);
            acc[m] = __builtin_amdgcn_wmma_f32_16x16x32_bf16(
                false, ahi.v, false, blo.v, (short)0, acc[m], false, false);
        }
    }

    // ---- Stage 3: fused epilogue: bias + exact GELU + dot with W2, cross-lane reduce
    const float b1c = b1[col];
    const float w2c = W2[col];
#pragma unroll
    for (int m = 0; m < 4; ++m) {
#pragma unroll
        for (int r = 0; r < 8; ++r) {
            float x = acc[m][r] + b1c;
            float g = 0.5f * x * (1.0f + erff(x * 0.70710678118654752f));
            float v = g * w2c;
            // butterfly within each 16-lane half (halves hold M=r and M=r+8)
            v += __shfl_xor(v, 1, 32);
            v += __shfl_xor(v, 2, 32);
            v += __shfl_xor(v, 4, 32);
            v += __shfl_xor(v, 8, 32);
            if (lane15 == 0) {
                atomicAdd(&partial[(m << 4) + r + (khalf << 3)], v);
            }
        }
    }
    __syncthreads();

    if (tid < MBLK) {
        out[row0 + tid] = partial[tid] + b2[0];
    }
}

extern "C" void kernel_launch(void* const* d_in, const int* in_sizes, int n_in,
                              void* d_out, int out_size, void* d_ws, size_t ws_size,
                              hipStream_t stream) {
    const float* emb = (const float*)d_in[0];
    const int*   sq1 = (const int*)d_in[1];
    const int*   sq2 = (const int*)d_in[2];
    const float* W1  = (const float*)d_in[3];
    const float* b1  = (const float*)d_in[4];
    const float* W2  = (const float*)d_in[5];
    const float* b2  = (const float*)d_in[6];
    float* out = (float*)d_out;

    unsigned short* Whi = (unsigned short*)d_ws;            // 256 KB
    unsigned short* Wlo = Whi + (size_t)KDIM * HDIM;        // 256 KB

    prep_w1_kernel<<<(KDIM * HDIM + 255) / 256, 256, 0, stream>>>(W1, Whi, Wlo);

    const size_t smemBytes = (size_t)2 * MBLK * RSTRIDE * sizeof(unsigned short)
                           + (size_t)MBLK * sizeof(float);  // 133,376 B
    const int nblocks = (BATCH * NMOVES) / MBLK;            // 7,680
    policy_head_kernel<<<nblocks, TPB, smemBytes, stream>>>(
        emb, sq1, sq2, Whi, Wlo, b1, W2, b2, out);
}
